// KernelMethod_65481071401189
// MI455X (gfx1250) — matrix operand order repaired and checked by reading, not compiled, not run on hardware
//
#include <hip/hip_runtime.h>

#define GAMMA 0.005f
#define NROWS 8192
#define DDIM  256
#define LDSTR 72   // 64 K-elements + 8 pad (bf16 units) to break bank conflicts

typedef __attribute__((ext_vector_type(16))) __bf16 v16bf;
typedef __attribute__((ext_vector_type(8)))  float  v8f;
typedef unsigned short u16;

union Frag {
    uint4 u[2];
    v16bf v;
};

static __device__ __forceinline__ void bf16_split(float f, u16& hi, u16& lo) {
    union { __bf16 b; u16 s; } c;
    __bf16 bh = (__bf16)f;           // RNE truncate to bf16
    float  fh = (float)bh;
    __bf16 bl = (__bf16)(f - fh);    // residual
    c.b = bh; hi = c.s;
    c.b = bl; lo = c.s;
}

// ---------------------------------------------------------------------------
// Pass 1: per-row squared norms of X and Y. One wave32 per row.
// ---------------------------------------------------------------------------
__global__ __launch_bounds__(256) void row_norms(const float* __restrict__ X,
                                                 const float* __restrict__ Y,
                                                 float* __restrict__ x2,
                                                 float* __restrict__ y2) {
    int lane = threadIdx.x & 31;
    int wid  = blockIdx.x * 8 + (threadIdx.x >> 5);   // 0 .. 16383
    const float* src;
    float* dst;
    int row;
    if (wid < NROWS) { src = X; dst = x2; row = wid; }
    else             { src = Y; dst = y2; row = wid - NROWS; }

    const float* p = src + (size_t)row * DDIM + lane * 8;
    float4 a = *(const float4*)(p);
    float4 b = *(const float4*)(p + 4);
    float s = a.x * a.x;
    s = fmaf(a.y, a.y, s); s = fmaf(a.z, a.z, s); s = fmaf(a.w, a.w, s);
    s = fmaf(b.x, b.x, s); s = fmaf(b.y, b.y, s);
    s = fmaf(b.z, b.z, s); s = fmaf(b.w, b.w, s);
    #pragma unroll
    for (int off = 16; off > 0; off >>= 1)
        s += __shfl_xor(s, off, 32);
    if (lane == 0) dst[row] = s;
}

// ---------------------------------------------------------------------------
// Pass 2: 128x128 output tile per block. Split-bf16 WMMA GEMM + exp epilogue.
//   out[i,j] = exp(2g * dot(x_i,y_j) - g*(|x_i|^2 + |y_j|^2))
// ---------------------------------------------------------------------------
__global__ __launch_bounds__(256) void rbf_gemm(const float* __restrict__ X,
                                                const float* __restrict__ Y,
                                                const float* __restrict__ x2,
                                                const float* __restrict__ y2,
                                                float* __restrict__ out) {
    __shared__ __align__(16) u16 sXhi[128 * LDSTR];
    __shared__ __align__(16) u16 sXlo[128 * LDSTR];
    __shared__ __align__(16) u16 sYhi[128 * LDSTR];
    __shared__ __align__(16) u16 sYlo[128 * LDSTR];

    const int t    = threadIdx.x;
    const int lane = t & 31;
    const int wave = t >> 5;
    const int wm   = wave & 3;     // wave row group (0..3) -> 32 rows each
    const int wn   = wave >> 2;    // wave col group (0..1) -> 64 cols each
    const int rowBlock = blockIdx.y * 128;
    const int colBlock = blockIdx.x * 128;

    const int halfSel = (lane >= 16) ? 1 : 0;
    const int l15     = lane & 15;

    v8f acc[2][4];
    v8f zero = {};
    #pragma unroll
    for (int mi = 0; mi < 2; ++mi)
        #pragma unroll
        for (int ni = 0; ni < 4; ++ni)
            acc[mi][ni] = zero;

    for (int chunk = 0; chunk < 4; ++chunk) {
        const int k0c = chunk * 64;
        __syncthreads();   // protect LDS from previous iteration's readers

        // Stage 128x64 f32 of X and Y, split into bf16 hi/lo in LDS.
        // 2048 float4 per tile -> 8 per thread.
        #pragma unroll
        for (int i = 0; i < 8; ++i) {
            int idx = t + i * 256;
            int r   = idx >> 4;             // 16 float4 per row of 64
            int c4  = (idx & 15) << 2;      // col within chunk
            int lo_off = r * LDSTR + c4;    // bf16-element offset in LDS

            float4 vx = *(const float4*)(X + (size_t)(rowBlock + r) * DDIM + k0c + c4);
            u16 h0, h1, h2, h3, q0, q1, q2, q3;
            bf16_split(vx.x, h0, q0); bf16_split(vx.y, h1, q1);
            bf16_split(vx.z, h2, q2); bf16_split(vx.w, h3, q3);
            uint2 hv, lv;
            hv.x = (uint)h0 | ((uint)h1 << 16); hv.y = (uint)h2 | ((uint)h3 << 16);
            lv.x = (uint)q0 | ((uint)q1 << 16); lv.y = (uint)q2 | ((uint)q3 << 16);
            *(uint2*)&sXhi[lo_off] = hv;
            *(uint2*)&sXlo[lo_off] = lv;

            float4 vy = *(const float4*)(Y + (size_t)(colBlock + r) * DDIM + k0c + c4);
            bf16_split(vy.x, h0, q0); bf16_split(vy.y, h1, q1);
            bf16_split(vy.z, h2, q2); bf16_split(vy.w, h3, q3);
            hv.x = (uint)h0 | ((uint)h1 << 16); hv.y = (uint)h2 | ((uint)h3 << 16);
            lv.x = (uint)q0 | ((uint)q1 << 16); lv.y = (uint)q2 | ((uint)q3 << 16);
            *(uint2*)&sYhi[lo_off] = hv;
            *(uint2*)&sYlo[lo_off] = lv;
        }
        __syncthreads();

        #pragma unroll
        for (int kk = 0; kk < 2; ++kk) {
            const int k0 = kk * 32;

            // A fragments (16x32 bf16): lanes 0-15 K 0..15, lanes 16-31 K 16..31 (contiguous)
            Frag ahi[2], alo[2];
            const int kA = k0 + (halfSel ? 16 : 0);
            #pragma unroll
            for (int mi = 0; mi < 2; ++mi) {
                int row = wm * 32 + mi * 16 + l15;
                int e   = row * LDSTR + kA;
                ahi[mi].u[0] = *(const uint4*)&sXhi[e];
                ahi[mi].u[1] = *(const uint4*)&sXhi[e + 8];
                alo[mi].u[0] = *(const uint4*)&sXlo[e];
                alo[mi].u[1] = *(const uint4*)&sXlo[e + 8];
            }

            // B fragments (32x16 bf16): lanes 0-15 K 0..15, lanes 16-31 K 16..31 (contiguous)
            Frag bhi[4], blo[4];
            const int kB = k0 + (halfSel ? 16 : 0);
            #pragma unroll
            for (int ni = 0; ni < 4; ++ni) {
                int yr = wn * 64 + ni * 16 + l15;
                int e  = yr * LDSTR + kB;
                bhi[ni].u[0] = *(const uint4*)&sYhi[e];
                bhi[ni].u[1] = *(const uint4*)&sYhi[e + 8];
                blo[ni].u[0] = *(const uint4*)&sYlo[e];
                blo[ni].u[1] = *(const uint4*)&sYlo[e + 8];
            }

            // Split-bf16 product: hi*hi + hi*lo + lo*hi
            #pragma unroll
            for (int mi = 0; mi < 2; ++mi) {
                #pragma unroll
                for (int ni = 0; ni < 4; ++ni) {
                    acc[mi][ni] = __builtin_amdgcn_wmma_f32_16x16x32_bf16(
                        false, ahi[mi].v, false, bhi[ni].v, (short)0, acc[mi][ni], false, false);
                    acc[mi][ni] = __builtin_amdgcn_wmma_f32_16x16x32_bf16(
                        false, ahi[mi].v, false, blo[ni].v, (short)0, acc[mi][ni], false, false);
                    acc[mi][ni] = __builtin_amdgcn_wmma_f32_16x16x32_bf16(
                        false, alo[mi].v, false, bhi[ni].v, (short)0, acc[mi][ni], false, false);
                }
            }
        }
    }

    // Epilogue: out = exp(2g*dot - g*(x2+y2))
    #pragma unroll
    for (int mi = 0; mi < 2; ++mi) {
        int rbase = rowBlock + wm * 32 + mi * 16 + (halfSel ? 8 : 0);
        #pragma unroll
        for (int ni = 0; ni < 4; ++ni) {
            int gc = colBlock + wn * 64 + ni * 16 + l15;
            float yv = y2[gc];
            #pragma unroll
            for (int r = 0; r < 8; ++r) {
                int gr = rbase + r;
                float arg = fmaf(2.0f * GAMMA, acc[mi][ni][r],
                                 -GAMMA * (x2[gr] + yv));
                out[(size_t)gr * NROWS + gc] = __expf(arg);
            }
        }
    }
}

extern "C" void kernel_launch(void* const* d_in, const int* in_sizes, int n_in,
                              void* d_out, int out_size, void* d_ws, size_t ws_size,
                              hipStream_t stream) {
    const float* x = (const float*)d_in[0];
    const float* y = (const float*)d_in[1];
    float* out = (float*)d_out;
    float* x2  = (float*)d_ws;          // 8192 floats
    float* y2  = x2 + NROWS;            // 8192 floats

    // 16384 rows total, one wave32 per row, 8 waves per 256-thread block.
    row_norms<<<dim3(2048), dim3(256), 0, stream>>>(x, y, x2, y2);
    // 64x64 blocks of 128x128 output tiles.
    rbf_gemm<<<dim3(64, 64), dim3(256), 0, stream>>>(x, y, x2, y2, out);
}
